// RNN_2156073583212
// MI455X (gfx1250) — matrix operand order repaired and checked by reading, not compiled, not run on hardware
//
#include <hip/hip_runtime.h>

typedef __attribute__((ext_vector_type(16))) _Float16 v16h;
typedef __attribute__((ext_vector_type(8)))  _Float16 v8h;
typedef __attribute__((ext_vector_type(8)))  float    v8f;

#define T_STEPS 512
#define HDIM    64
#define HSTR    72   // padded LDS row stride in f16 elems (72*2B=144B -> conflict-free b128)
#define WAVES_PER_BLOCK 2
#define ROWS_PER_WAVE   16

__global__ __launch_bounds__(32 * WAVES_PER_BLOCK)
void rnn_wmma_f16(const float* __restrict__ x,     // (B, T)
                  const float* __restrict__ W_xh,  // (64, 1)
                  const float* __restrict__ W_hh,  // (64, 64)
                  const float* __restrict__ W_hy,  // (1, 64)
                  float* __restrict__ out)         // (B,)
{
    __shared__ __align__(16) _Float16 lds_h[WAVES_PER_BLOCK * 16 * HSTR];

    const int tid  = threadIdx.x;
    const int wave = tid >> 5;
    const int lane = tid & 31;
    const int half = lane >> 4;   // 0: lanes 0-15, 1: lanes 16-31
    const int lm   = lane & 15;

    _Float16* hbuf = &lds_h[wave * 16 * HSTR];

    const int b0 = blockIdx.x * (WAVES_PER_BLOCK * ROWS_PER_WAVE) + wave * ROWS_PER_WAVE;
    const float* __restrict__ xrow = x + (size_t)(b0 + lm) * T_STEPS;

    // ---- Preload A operands: W_hh in WMMA 16x32 f16 A-layout, kept in VGPRs all 512 steps.
    // A-layout (16-bit, 16x32): lanes 0-15 row M=lm hold K = 0..7 (elems 0..7) and
    // K = 16..23 (elems 8..15); lanes 16-31 hold K = 8..15 and 24..31.
    v16h Aop[4][2];
    #pragma unroll
    for (int m = 0; m < 4; ++m) {
      #pragma unroll
      for (int kc = 0; kc < 2; ++kc) {
        const int row = m * 16 + lm;           // output hidden index i
        const int kb  = kc * 32 + half * 8;    // K base within 64
        v16h a;
        #pragma unroll
        for (int p = 0; p < 8; ++p) {
          a[p]     = (_Float16)W_hh[row * HDIM + kb + p];
          a[8 + p] = (_Float16)W_hh[row * HDIM + kb + 16 + p];
        }
        Aop[m][kc] = a;
      }
    }

    // ---- Bias weights w_x in C/D layout: tile m, VGPR v holds i = m*16 + half*8 + v.
    float wxr[4][8];
    #pragma unroll
    for (int m = 0; m < 4; ++m)
      #pragma unroll
      for (int v = 0; v < 8; ++v)
        wxr[m][v] = W_xh[m * 16 + half * 8 + v];

    // ---- t = 0: h0 = 0  =>  pre-relu h = x_0 * w_x   (relu is deferred to the f16 pack)
    v8f acc[4];
    {
      const float x0 = xrow[0];
      #pragma unroll
      for (int m = 0; m < 4; ++m)
        #pragma unroll
        for (int v = 0; v < 8; ++v)
          acc[m][v] = x0 * wxr[m][v];
    }

    const v8h zero8 = { (_Float16)0.f, (_Float16)0.f, (_Float16)0.f, (_Float16)0.f,
                       (_Float16)0.f, (_Float16)0.f, (_Float16)0.f, (_Float16)0.f };

    // ---- Recurrence: 511 more steps. h'(64x16) = relu(W_hh @ h' + x_t (x) w_x)
    // acc holds the PRE-relu step output; relu is applied in the f16 domain during the
    // pack (relu(cvt(x)) == cvt(relu(x)) under RTNE), so it costs 4 v_pk_max_num_f16
    // per tile instead of 2x v_max_num_f32 per element.
    for (int t = 1; t < T_STEPS; ++t) {
      // Store h (relu'd, f16) into LDS, layout hbuf[batch][hidden].
      // D-tile m, VGPR v (this lane) = h[i = m*16 + half*8 + v][b = lm] -> contiguous 8 f16.
      #pragma unroll
      for (int m = 0; m < 4; ++m) {
        v8h hv = __builtin_convertvector(acc[m], v8h);    // v_cvt (canonical results)
        hv = __builtin_elementwise_max(hv, zero8);        // v_pk_max_num_f16, no canon max
        *(v8h*)(&hbuf[lm * HSTR + m * 16 + half * 8]) = hv;  // ds_store_b128, 16B aligned
      }
      __asm__ volatile("" ::: "memory");  // DS is in-order per wave; just pin compiler order

      // Rebuild B operands (32x16 f16 B-layout): lane lm = N(batch),
      // lanes 0-15 hold K = kbase+0..7,16..23, lanes 16-31 hold K = kbase+8..15,24..31.
      v16h Bop[2];
      #pragma unroll
      for (int c = 0; c < 2; ++c) {
        const int col = c * 32 + half * 8;
        v8h p0 = *(const v8h*)(&hbuf[lm * HSTR + col]);      // K = col .. col+7
        v8h p1 = *(const v8h*)(&hbuf[lm * HSTR + col + 16]); // K = col+16 .. col+23
        Bop[c] = __builtin_shufflevector(p0, p1, 0,1,2,3,4,5,6,7,8,9,10,11,12,13,14,15);
      }
      __asm__ volatile("" ::: "memory");

      const float xv = xrow[t];  // x row is L2/L0 resident; 32 steps share a cacheline

      #pragma unroll
      for (int m = 0; m < 4; ++m) {
        v8f c;
        #pragma unroll
        for (int v = 0; v < 8; ++v) c[v] = xv * wxr[m][v];   // bias as WMMA C operand
        c = __builtin_amdgcn_wmma_f32_16x16x32_f16(false, Aop[m][0], false, Bop[0],
                                                   (short)0, c, false, false);
        c = __builtin_amdgcn_wmma_f32_16x16x32_f16(false, Aop[m][1], false, Bop[1],
                                                   (short)0, c, false, false);
        acc[m] = c;                                          // pre-relu; relu at pack
      }
    }

    // ---- Output projection: out[b] = sum_i relu(h_pre[b][i]) * W_hy[i]
    // (apply the deferred relu of the final step here, once, in f32)
    float sum = 0.0f;
    #pragma unroll
    for (int m = 0; m < 4; ++m)
      #pragma unroll
      for (int v = 0; v < 8; ++v)
        sum += fmaxf(acc[m][v], 0.0f) * W_hy[m * 16 + half * 8 + v];
    // lanes L and L+16 hold complementary halves of i for the same batch b = lm
    sum += __shfl_down(sum, 16, 32);
    if (half == 0) out[b0 + lm] = sum;
}

extern "C" void kernel_launch(void* const* d_in, const int* in_sizes, int n_in,
                              void* d_out, int out_size, void* d_ws, size_t ws_size,
                              hipStream_t stream) {
    const float* x    = (const float*)d_in[0];  // (B, T) f32
    const float* W_xh = (const float*)d_in[1];  // (64, 1)
    const float* W_hh = (const float*)d_in[2];  // (64, 64)
    const float* W_hy = (const float*)d_in[3];  // (1, 64)
    float* out = (float*)d_out;

    const int B = in_sizes[0] / T_STEPS;                        // 8192
    const int rows_per_block = WAVES_PER_BLOCK * ROWS_PER_WAVE; // 32
    dim3 grid(B / rows_per_block);                              // 256 blocks
    dim3 block(32 * WAVES_PER_BLOCK);                           // 64 threads = 2 waves
    rnn_wmma_f16<<<grid, block, 0, stream>>>(x, W_xh, W_hh, W_hy, out);
}
